// GAT_20194936226187
// MI455X (gfx1250) — matrix operand order repaired and checked by reading, not compiled, not run on hardware
//
#include <hip/hip_runtime.h>
#include <hip/hip_bf16.h>
#include <stdint.h>

// ---------------- problem constants (from reference setup_inputs) -----------
#define NN   20000   // nodes
#define FF   128     // input features
#define EE   320000  // edges
#define D0   256     // E0 hidden
#define D1   512     // E1 hidden (= H*C)
#define HH   2       // heads
#define CC   256     // per-head channels
#define GG   128     // graphs
#define NCLS 10      // classes
#define BN_EPS 1e-5f

typedef __bf16 bf16_t;
typedef __attribute__((ext_vector_type(16))) __bf16 v16bf;
typedef __attribute__((ext_vector_type(8)))  float  v8f;

union Frag16 { v16bf v; uint4 q[2]; };

__device__ __forceinline__ unsigned fkey(float f) {
  unsigned u = __float_as_uint(f);
  return (u & 0x80000000u) ? ~u : (u | 0x80000000u);
}
__device__ __forceinline__ float fkey_dec(unsigned k) {
  return (k & 0x80000000u) ? __uint_as_float(k & 0x7FFFFFFFu)
                           : __uint_as_float(~k);
}

// ---------------- conversion / packing --------------------------------------
__global__ void k_f32_to_bf16(const float* __restrict__ in, bf16_t* __restrict__ out, int n) {
  int i = blockIdx.x * blockDim.x + threadIdx.x;
  if (i < n) out[i] = (bf16_t)in[i];
}

// Pack W (K x Nc, row-major f32) into WMMA B-fragment order, grouped so the
// 4 fragments of one 64-wide N-group at the same k-step are contiguous:
//   frag index = (ngroup*KT + kt)*4 + j   (ntile = ngroup*4 + j)
// within fragment: 32 lanes x 16 elems; lane n = lane&15, khalf = lane>>4,
// elem e -> k = kt*32 + khalf*8 + (e&7) + (e>>3)*16.
__global__ void k_pack_b(const float* __restrict__ W, bf16_t* __restrict__ out, int K, int Nc) {
  int i = blockIdx.x * blockDim.x + threadIdx.x;
  if (i >= K * Nc) return;
  int KT   = K >> 5;
  int frag = i >> 9;        // 512 elems per fragment
  int r    = i & 511;
  int lane = r >> 4;
  int e    = r & 15;
  int j   = frag & 3;
  int fg  = frag >> 2;
  int ng  = fg / KT;
  int kt  = fg - ng * KT;
  int ntile = ng * 4 + j;
  int n = (ntile << 4) + (lane & 15);
  int k = (kt << 5) + ((lane >> 4) << 3) + (e & 7) + ((e >> 3) << 4);
  out[i] = (bf16_t)W[(size_t)k * Nc + n];
}

// ---------------- WMMA GEMM helpers -----------------------------------------
__device__ __forceinline__ void lda(Frag16& a, const bf16_t* p) {
  a.q[0] = *(const uint4*)(p);
  a.q[1] = *(const uint4*)(p + 16);
}
__device__ __forceinline__ void ldb4(Frag16 b[4], const bf16_t* p) {
#pragma unroll
  for (int j = 0; j < 4; ++j) {
    b[j].q[0] = *(const uint4*)(p + j * 512);
    b[j].q[1] = *(const uint4*)(p + j * 512 + 8);
  }
}
__device__ __forceinline__ void mma4(v8f acc[4], const Frag16& a, const Frag16 b[4]) {
#pragma unroll
  for (int j = 0; j < 4; ++j)
    acc[j] = __builtin_amdgcn_wmma_f32_16x16x32_bf16(
        false, a.v, false, b[j].v, (short)0, acc[j], false, false);
}

// ---------------- WMMA GEMM: C = A(MxK) * Bp(KxNc packed) -------------------
// Each wave computes a 16x64 strip (4 WMMA tiles) with a shared A fragment and
// double-buffered operands (prefetch k-step kt+1 before issuing WMMAs for kt)
// so global loads overlap the matrix pipe. Requires KT even, KT >= 4.
// MODE 0: store f32 C (no bias)        -> Cf
// MODE 1: y = relu(bn(C + bias)) bf16  -> Cb
template <int MODE>
__global__ __launch_bounds__(128)
void k_gemm_wmma(const bf16_t* __restrict__ A, const bf16_t* __restrict__ Bp,
                 int M, int K, int Nc,
                 const float* __restrict__ bias, const float* __restrict__ bg,
                 const float* __restrict__ bb,   const float* __restrict__ bm,
                 const float* __restrict__ bv,
                 float* __restrict__ Cf, bf16_t* __restrict__ Cb) {
  const int lane = threadIdx.x;                     // 0..31 (wave32)
  const int ntg  = blockIdx.x * 4 + threadIdx.y;    // 64-wide N group index
  const int m0   = blockIdx.y << 4;
  const int hi   = lane >> 4;
  const int lo   = lane & 15;
  const int KT   = K >> 5;

  const bf16_t* arow  = A + (size_t)(m0 + lo) * K + hi * 8;
  const bf16_t* bbase = Bp + ((size_t)ntg * KT << 11) + (lane << 4); // KT*2048 elems/group

  v8f acc[4] = {};
  Frag16 a0, a1;
  Frag16 b0[4], b1[4];

  lda(a0, arow);                 // kt = 0
  ldb4(b0, bbase);
  for (int kt = 0; kt < KT - 2; kt += 2) {
    lda(a1, arow + 32);          // prefetch kt+1
    ldb4(b1, bbase + 2048);
    mma4(acc, a0, b0);           // compute kt
    arow  += 64;
    bbase += 4096;
    lda(a0, arow);               // prefetch kt+2
    ldb4(b0, bbase);
    mma4(acc, a1, b1);           // compute kt+1
  }
  lda(a1, arow + 32);            // kt = KT-1
  ldb4(b1, bbase + 2048);
  mma4(acc, a0, b0);             // kt = KT-2
  mma4(acc, a1, b1);             // kt = KT-1

#pragma unroll
  for (int j = 0; j < 4; ++j) {
    const int n = ((ntg * 4 + j) << 4) + lo;
    if constexpr (MODE == 1) {
      float sA = bg[n] * rsqrtf(bv[n] + BN_EPS);
      float sB = (bias[n] - bm[n]) * sA + bb[n];
#pragma unroll
      for (int r = 0; r < 8; ++r) {
        int row = m0 + r + (hi << 3);
        float y = acc[j][r] * sA + sB;
        Cb[(size_t)row * Nc + n] = (bf16_t)fmaxf(y, 0.f);
      }
    } else {
#pragma unroll
      for (int r = 0; r < 8; ++r) {
        int row = m0 + r + (hi << 3);
        Cf[(size_t)row * Nc + n] = acc[j][r];
      }
    }
  }
}

// ---------------- GAT attention pieces (fp32) -------------------------------
// one wave per node: ls/ld[n,h] = dot(xw[n,h,:], a_src/dst[h,:])
__global__ void k_gat_logits(const float* __restrict__ xw,
                             const float* __restrict__ asrc,
                             const float* __restrict__ adst,
                             float* __restrict__ ls, float* __restrict__ ld) {
  int wv = threadIdx.x >> 5, lane = threadIdx.x & 31;
  int n = blockIdx.x * 4 + wv;
  if (n >= NN) return;
  for (int h = 0; h < HH; ++h) {
    const float* row = xw + (size_t)n * D1 + h * CC;
    float s = 0.f, d = 0.f;
    for (int c = lane; c < CC; c += 32) {
      float xv = row[c];
      s += xv * asrc[h * CC + c];
      d += xv * adst[h * CC + c];
    }
    for (int off = 16; off; off >>= 1) {
      s += __shfl_xor(s, off, 32);
      d += __shfl_xor(d, off, 32);
    }
    if (lane == 0) { ls[n * HH + h] = s; ld[n * HH + h] = d; }
  }
}

__global__ void k_edge_logit_max(const int* __restrict__ ei, const int* __restrict__ ew,
                                 const float* __restrict__ ls, const float* __restrict__ ld,
                                 float* __restrict__ ebuf, unsigned* __restrict__ mkey) {
  int idx = blockIdx.x * blockDim.x + threadIdx.x;
  if (idx >= EE * HH) return;
  int e = idx >> 1, h = idx & 1;
  int s = ei[e], d = ei[EE + e];
  float v = ls[s * HH + h] + ld[d * HH + h];
  v = (v > 0.f) ? v : 0.2f * v;          // LeakyReLU(0.2)
  if (ew[e] != 1) v = -1e30f;            // masked edge
  ebuf[idx] = v;
  atomicMax(&mkey[d * HH + h], fkey(v)); // segment max (ordered-uint key)
}

__global__ void k_edge_exp(const int* __restrict__ ei, const int* __restrict__ ew,
                           float* __restrict__ ebuf, const unsigned* __restrict__ mkey,
                           float* __restrict__ den) {
  int idx = blockIdx.x * blockDim.x + threadIdx.x;
  if (idx >= EE * HH) return;
  int e = idx >> 1, h = idx & 1;
  int d = ei[EE + e];
  float m = fmaxf(fkey_dec(mkey[d * HH + h]), -1e30f);
  float p = (ew[e] == 1) ? expf(ebuf[idx] - m) : 0.f;
  ebuf[idx] = p;                          // reuse buffer: logits -> p
  if (p != 0.f) atomicAdd(&den[d * HH + h], p);
}

// one block per edge, 128 threads x float4 = 512 features
__global__ void k_edge_scatter(const int* __restrict__ ei, const float* __restrict__ p,
                               const float* __restrict__ den, const float* __restrict__ xw,
                               float* __restrict__ agg) {
  int e = blockIdx.x;
  int s = ei[e], d = ei[EE + e];
  int c0 = threadIdx.x * 4;
  int h  = c0 >> 8;
  float alpha = p[e * HH + h] / (den[d * HH + h] + 1e-16f);
  if (alpha != 0.f) {
    float4 xv = *(const float4*)(xw + (size_t)s * D1 + c0);
    float* o = agg + (size_t)d * D1 + c0;
    atomicAdd(o + 0, alpha * xv.x);
    atomicAdd(o + 1, alpha * xv.y);
    atomicAdd(o + 2, alpha * xv.z);
    atomicAdd(o + 3, alpha * xv.w);
  }
}

// h3 = bf16(agg + bias)  (feeds next WMMA GEMM)
__global__ void k_bias_bf16(const float* __restrict__ agg, const float* __restrict__ bias,
                            bf16_t* __restrict__ out, int n) {
  int i = blockIdx.x * blockDim.x + threadIdx.x;
  if (i < n) out[i] = (bf16_t)(agg[i] + bias[i & (D1 - 1)]);
}

// ---------------- pooling + head --------------------------------------------
__global__ void k_pool(const float* __restrict__ agg, const float* __restrict__ bias,
                       const int* __restrict__ batch,
                       float* __restrict__ pooled, float* __restrict__ cnt) {
  int n = blockIdx.x;
  int g = batch[n];
  for (int f = threadIdx.x; f < D1; f += blockDim.x)
    atomicAdd(&pooled[(size_t)g * D1 + f], agg[(size_t)n * D1 + f] + bias[f]);
  if (threadIdx.x == 0) atomicAdd(&cnt[g], 1.0f);
}

__global__ void k_head_bn(const float* __restrict__ pooled, const float* __restrict__ cnt,
                          const float* __restrict__ g_, const float* __restrict__ b_,
                          const float* __restrict__ m_, const float* __restrict__ v_,
                          float* __restrict__ z) {
  int i = blockIdx.x * blockDim.x + threadIdx.x;
  if (i >= GG * D1) return;
  int f = i & (D1 - 1), gr = i >> 9;
  float c = fmaxf(cnt[gr], 1.0f);
  float x = pooled[i] / c;
  float y = (x - m_[f]) * rsqrtf(v_[f] + BN_EPS) * g_[f] + b_[f];
  z[i] = fmaxf(y, 0.f);
}

__global__ void k_head_fc1(const float* __restrict__ z, const float* __restrict__ w,
                           const float* __restrict__ b, float* __restrict__ z1) {
  int i = blockIdx.x * blockDim.x + threadIdx.x;
  if (i >= GG * D1) return;
  int j = i & (D1 - 1), gr = i >> 9;
  float acc = b[j];
  for (int k = 0; k < D1; ++k) acc += z[gr * D1 + k] * w[(size_t)k * D1 + j];
  z1[i] = fmaxf(acc, 0.f);
}

__global__ void k_head_fc2(const float* __restrict__ z1, const float* __restrict__ w,
                           const float* __restrict__ b, float* __restrict__ out) {
  int i = blockIdx.x * blockDim.x + threadIdx.x;
  if (i >= GG * NCLS) return;
  int j = i % NCLS, gr = i / NCLS;
  float acc = b[j];
  for (int k = 0; k < D1; ++k) acc += z1[gr * D1 + k] * w[(size_t)k * NCLS + j];
  out[i] = acc;
}

// ---------------- host orchestration ----------------------------------------
#define CDIV(a, b) (((a) + (b) - 1) / (b))

extern "C" void kernel_launch(void* const* d_in, const int* in_sizes, int n_in,
                              void* d_out, int out_size, void* d_ws, size_t ws_size,
                              hipStream_t stream) {
  (void)in_sizes; (void)n_in; (void)out_size; (void)ws_size;
  const float* x    = (const float*)d_in[0];
  const int*   ei   = (const int*)d_in[1];
  const int*   ew   = (const int*)d_in[2];
  const int*   bat  = (const int*)d_in[3];
  const float* w1   = (const float*)d_in[4];
  const float* b1   = (const float*)d_in[5];
  const float *bn1g = (const float*)d_in[6],  *bn1b = (const float*)d_in[7],
              *bn1m = (const float*)d_in[8],  *bn1v = (const float*)d_in[9];
  const float* w2   = (const float*)d_in[10];
  const float* b2   = (const float*)d_in[11];
  const float *bn2g = (const float*)d_in[12], *bn2b = (const float*)d_in[13],
              *bn2m = (const float*)d_in[14], *bn2v = (const float*)d_in[15];
  const float* g0w  = (const float*)d_in[16];
  const float* g0as = (const float*)d_in[17];
  const float* g0ad = (const float*)d_in[18];
  const float* g0b  = (const float*)d_in[19];
  const float* g1w  = (const float*)d_in[20];
  const float* g1as = (const float*)d_in[21];
  const float* g1ad = (const float*)d_in[22];
  const float* g1b  = (const float*)d_in[23];
  const float *bn3g = (const float*)d_in[24], *bn3b = (const float*)d_in[25],
              *bn3m = (const float*)d_in[26], *bn3v = (const float*)d_in[27];
  const float* f1w  = (const float*)d_in[28];
  const float* f1b  = (const float*)d_in[29];
  const float* f2w  = (const float*)d_in[30];
  const float* f2b  = (const float*)d_in[31];
  float* out = (float*)d_out;

  // ---- carve workspace (256B aligned) ----
  char* base = (char*)d_ws;
  size_t off = 0;
  auto alloc = [&](size_t bytes) -> void* {
    void* r = base + off;
    off = (off + bytes + 255) & ~(size_t)255;
    return r;
  };
  bf16_t* xb   = (bf16_t*)alloc((size_t)NN * FF * 2);
  bf16_t* w1p  = (bf16_t*)alloc((size_t)FF * D0 * 2);
  bf16_t* w2p  = (bf16_t*)alloc((size_t)D0 * D0 * 2);
  bf16_t* g0wp = (bf16_t*)alloc((size_t)D0 * D1 * 2);
  bf16_t* g1wp = (bf16_t*)alloc((size_t)D1 * D1 * 2);
  bf16_t* h1b  = (bf16_t*)alloc((size_t)NN * D0 * 2);
  bf16_t* h2b  = (bf16_t*)alloc((size_t)NN * D0 * 2);
  bf16_t* h3b  = (bf16_t*)alloc((size_t)NN * D1 * 2);
  float*  xw   = (float*) alloc((size_t)NN * D1 * 4);   // reused: xw0 then xw1
  float*  agg  = (float*) alloc((size_t)NN * D1 * 4);   // reused: agg0 then agg1
  float*  ls   = (float*) alloc((size_t)NN * HH * 4);
  float*  ld   = (float*) alloc((size_t)NN * HH * 4);
  unsigned* mk = (unsigned*)alloc((size_t)NN * HH * 4);
  float*  den  = (float*) alloc((size_t)NN * HH * 4);
  float*  ebuf = (float*) alloc((size_t)EE * HH * 4);   // logits -> p in place
  float*  pool = (float*) alloc((size_t)GG * D1 * 4);
  float*  cnt  = (float*) alloc((size_t)GG * 4);
  float*  z    = (float*) alloc((size_t)GG * D1 * 4);
  float*  z1   = (float*) alloc((size_t)GG * D1 * 4);

  dim3 gblk(32, 4);  // 4 waves per block; each wave: 16x64 strip (4 WMMA tiles)

  // ---- prepare bf16 operands ----
  k_f32_to_bf16<<<CDIV(NN * FF, 256), 256, 0, stream>>>(x, xb, NN * FF);
  k_pack_b<<<CDIV(FF * D0, 256), 256, 0, stream>>>(w1, w1p, FF, D0);
  k_pack_b<<<CDIV(D0 * D0, 256), 256, 0, stream>>>(w2, w2p, D0, D0);
  k_pack_b<<<CDIV(D0 * D1, 256), 256, 0, stream>>>(g0w, g0wp, D0, D1);
  k_pack_b<<<CDIV(D1 * D1, 256), 256, 0, stream>>>(g1w, g1wp, D1, D1);

  // ---- initial MLP (GEMM + BN + ReLU fused, WMMA) ----
  k_gemm_wmma<1><<<dim3(D0 / 256, NN / 16), gblk, 0, stream>>>(
      xb, w1p, NN, FF, D0, b1, bn1g, bn1b, bn1m, bn1v, nullptr, h1b);
  k_gemm_wmma<1><<<dim3(D0 / 256, NN / 16), gblk, 0, stream>>>(
      h1b, w2p, NN, D0, D0, b2, bn2g, bn2b, bn2m, bn2v, nullptr, h2b);

  // ---- GAT layer 0 ----
  k_gemm_wmma<0><<<dim3(D1 / 256, NN / 16), gblk, 0, stream>>>(
      h2b, g0wp, NN, D0, D1, nullptr, nullptr, nullptr, nullptr, nullptr, xw, nullptr);
  k_gat_logits<<<CDIV(NN, 4), 128, 0, stream>>>(xw, g0as, g0ad, ls, ld);
  hipMemsetAsync(mk, 0, (size_t)NN * HH * 4, stream);
  hipMemsetAsync(den, 0, (size_t)NN * HH * 4, stream);
  hipMemsetAsync(agg, 0, (size_t)NN * D1 * 4, stream);
  k_edge_logit_max<<<CDIV(EE * HH, 256), 256, 0, stream>>>(ei, ew, ls, ld, ebuf, mk);
  k_edge_exp<<<CDIV(EE * HH, 256), 256, 0, stream>>>(ei, ew, ebuf, mk, den);
  k_edge_scatter<<<EE, 128, 0, stream>>>(ei, ebuf, den, xw, agg);
  k_bias_bf16<<<CDIV(NN * D1, 256), 256, 0, stream>>>(agg, g0b, h3b, NN * D1);

  // ---- GAT layer 1 ----
  k_gemm_wmma<0><<<dim3(D1 / 256, NN / 16), gblk, 0, stream>>>(
      h3b, g1wp, NN, D1, D1, nullptr, nullptr, nullptr, nullptr, nullptr, xw, nullptr);
  k_gat_logits<<<CDIV(NN, 4), 128, 0, stream>>>(xw, g1as, g1ad, ls, ld);
  hipMemsetAsync(mk, 0, (size_t)NN * HH * 4, stream);
  hipMemsetAsync(den, 0, (size_t)NN * HH * 4, stream);
  hipMemsetAsync(agg, 0, (size_t)NN * D1 * 4, stream);
  k_edge_logit_max<<<CDIV(EE * HH, 256), 256, 0, stream>>>(ei, ew, ls, ld, ebuf, mk);
  k_edge_exp<<<CDIV(EE * HH, 256), 256, 0, stream>>>(ei, ew, ebuf, mk, den);
  k_edge_scatter<<<EE, 128, 0, stream>>>(ei, ebuf, den, xw, agg);

  // ---- global mean pool + head ----
  hipMemsetAsync(pool, 0, (size_t)GG * D1 * 4, stream);
  hipMemsetAsync(cnt, 0, (size_t)GG * 4, stream);
  k_pool<<<NN, 256, 0, stream>>>(agg, g1b, bat, pool, cnt);
  k_head_bn<<<CDIV(GG * D1, 256), 256, 0, stream>>>(pool, cnt, bn3g, bn3b, bn3m, bn3v, z);
  k_head_fc1<<<CDIV(GG * D1, 256), 256, 0, stream>>>(z, f1w, f1b, z1);
  k_head_fc2<<<CDIV(GG * NCLS, 64), 64, 0, stream>>>(z1, f2w, f2b, out);
}
